// DHGLayer_23648089932276
// MI455X (gfx1250) — matrix operand order repaired and checked by reading, not compiled, not run on hardware
//
#include <hip/hip_runtime.h>
#include <math.h>

// ---------------------------------------------------------------------------
// DHG layer for MI455X (gfx1250, wave32).
// GEMMs via v_wmma_f32_16x16x32_bf16; similarity B-tiles staged into LDS with
// the Tensor Data Mover (tensor_load_to_lds + s_wait_tensorcnt) when the
// toolchain exposes it, else async global->LDS loads, else plain loads.
// This toolchain exposes the 6-arg TDM builtin (clang-23 / therock headers):
//   (uint32x4 g0, int32x8 g1, int32x4 g2, int32x4 g3, int32x8 g4, i32 cpol)
// ---------------------------------------------------------------------------

typedef __attribute__((ext_vector_type(16))) __bf16 v16bf;
typedef __attribute__((ext_vector_type(8)))  float  v8f;
typedef __attribute__((ext_vector_type(4)))  unsigned int u32x4;
typedef __attribute__((ext_vector_type(8)))  int i32x8;
typedef __attribute__((ext_vector_type(4)))  int i32x4;

#define NV   10000
#define DD   128
#define KS_  32
#define KN_  16
#define KC_  16

#if defined(__has_builtin)
#if __has_builtin(__builtin_amdgcn_tensor_load_to_lds) && \
    __has_builtin(__builtin_amdgcn_s_wait_tensorcnt)
#define USE_TDM 1
#elif __has_builtin(__builtin_amdgcn_global_load_async_to_lds_b128)
#define USE_ASYNC 1
#endif
#endif

// ---- WMMA fragment loaders (wave32, K = k0 + 16*(lane>>4) + e for A and B) --------------

// A 16x32 bf16 chunk at K-offset k0. `row` = this lane's A-row (M = lane&15).
__device__ __forceinline__ v16bf load_a_frag(const __bf16* row, int k0, int hi) {
  v16bf a;
#pragma unroll
  for (int e = 0; e < 16; ++e) a[e] = row[k0 + hi * 16 + e];
  return a;
}

// B 32x16 bf16 chunk at K-offset k0. `colrow` = source row holding column
// (lane&15)'s K-vector contiguously. lanes0-15 K=k0.., lanes16-31 K=k0+16..
__device__ __forceinline__ v16bf load_b_frag(const __bf16* colrow, int k0, int hi) {
  const __bf16* p = colrow + k0 + hi * 16;
  v16bf b;
#pragma unroll
  for (int e = 0; e < 16; ++e) b[e] = p[e];
  return b;
}

// ---- prep: row norms + bf16 copies ----------------------------------------

__global__ void __launch_bounds__(256) prep_norm_kernel(
    const float* __restrict__ feats, __bf16* __restrict__ xn_bf,
    __bf16* __restrict__ f_bf, int n_total) {
  int lane = threadIdx.x & 31, wid = threadIdx.x >> 5;
  int row = blockIdx.x * 8 + wid;
  if (row >= n_total) return;
  const float* fr = feats + (size_t)row * DD;
  float v0 = fr[lane], v1 = fr[lane + 32], v2 = fr[lane + 64], v3 = fr[lane + 96];
  float s = v0 * v0 + v1 * v1 + v2 * v2 + v3 * v3;
#pragma unroll
  for (int off = 16; off; off >>= 1) s += __shfl_xor(s, off, 32);
  float inv = 1.f / fmaxf(sqrtf(s), 1e-12f);
  size_t base = (size_t)row * DD;
  f_bf[base + lane]      = (__bf16)v0;
  f_bf[base + lane + 32] = (__bf16)v1;
  f_bf[base + lane + 64] = (__bf16)v2;
  f_bf[base + lane + 96] = (__bf16)v3;
  xn_bf[base + lane]      = (__bf16)(v0 * inv);
  xn_bf[base + lane + 32] = (__bf16)(v1 * inv);
  xn_bf[base + lane + 64] = (__bf16)(v2 * inv);
  xn_bf[base + lane + 96] = (__bf16)(v3 * inv);
}

__global__ void prep_weights_kernel(
    const float* __restrict__ wc, const float* __restrict__ wn,
    const float* __restrict__ wst, const float* __restrict__ fcw,
    __bf16* __restrict__ wc_bf, __bf16* __restrict__ wn_bf,
    __bf16* __restrict__ ws_bf, __bf16* __restrict__ fcwT_bf) {
  int tid = blockIdx.x * blockDim.x + threadIdx.x;
  int stride = gridDim.x * blockDim.x;
  for (int i = tid; i < KC_ * KC_ * DD; i += stride) wc_bf[i] = (__bf16)wc[i];
  for (int i = tid; i < KN_ * KN_ * DD; i += stride) wn_bf[i] = (__bf16)wn[i];
  for (int i = tid; i < KS_ * KS_ * DD; i += stride) ws_bf[i] = (__bf16)wst[i];
  for (int i = tid; i < DD * DD; i += stride)
    fcwT_bf[(size_t)(i & (DD - 1)) * DD + (i >> 7)] = (__bf16)fcw[i];
}

// ---- B-tile staging: global (16 rows x 128 bf16, row-major) -> LDS --------

__device__ __forceinline__ void stage_tile(const __bf16* __restrict__ xn,
                                           int col0, __bf16* dstbuf, int lane) {
#if defined(USE_TDM)
  // Tensor Data Mover: 2D tile, data_size=2B, tile 128x16, stride 128.
  // D# layout per CDNA5 ISA. Wave-uniform -> SGPRs.
  unsigned lds_off =
      (unsigned)(size_t)(__attribute__((address_space(3))) __bf16*)dstbuf;
  unsigned long long ga = (unsigned long long)(size_t)(xn + (size_t)col0 * DD);
  u32x4 g0;
  g0[0] = 1u;                                   // count=1, user descriptor
  g0[1] = lds_off;                              // lds_addr (bytes)
  g0[2] = (unsigned)ga;                         // global_addr[31:0]
  g0[3] = (unsigned)((ga >> 32) & 0x1FFFFFFu)   // global_addr[56:32]
          | (2u << 30);                         // type=2 ("image")
  i32x8 g1;
  g1[0] = (int)(1u << 16);                      // data_size=1 -> 2 bytes
  g1[1] = (int)(128u << 16);                    // tensor_dim0 = 128
  g1[2] = (int)(16u << 16);                     // tensor_dim1 = 16
  g1[3] = (int)(128u << 16);                    // tile_dim0 = 128
  g1[4] = 16;                                   // tile_dim1 = 16 (tile_dim2=0)
  g1[5] = 128;                                  // tensor_dim0_stride = 128
  g1[6] = 0;
  g1[7] = 0;
  i32x4 g2 = {0, 0, 0, 0}, g3 = {0, 0, 0, 0};   // <=2D tensor
  i32x8 g4 = {0, 0, 0, 0, 0, 0, 0, 0};          // extra group (zero-filled)
  __builtin_amdgcn_tensor_load_to_lds(g0, g1, g2, g3, g4, 0);
#elif defined(USE_ASYNC)
  const char* g = (const char*)(xn + (size_t)col0 * DD);
  char* l = (char*)dstbuf;
#pragma unroll
  for (int it = 0; it < 8; ++it) {              // 8 x (32 lanes x 16B) = 4KB
    int off = it * 512 + lane * 16;
    __builtin_amdgcn_global_load_async_to_lds_b128(
        (const __attribute__((address_space(1))) void*)(g + off),
        (__attribute__((address_space(3))) void*)(l + off), 0, 0);
  }
#else
  const char* g = (const char*)(xn + (size_t)col0 * DD);
  char* l = (char*)dstbuf;
#pragma unroll
  for (int it = 0; it < 8; ++it) {
    int off = it * 512 + lane * 16;
    *(int4*)(l + off) = *(const int4*)(g + off);
  }
#endif
}

__device__ __forceinline__ void stage_wait() {
#if defined(USE_TDM)
  __builtin_amdgcn_s_wait_tensorcnt(1);   // 1 outstanding (the next tile)
#elif defined(USE_ASYNC)
#if __has_builtin(__builtin_amdgcn_s_wait_asynccnt)
  __builtin_amdgcn_s_wait_asynccnt(8);    // next tile's 8 loads may remain
#else
  asm volatile("s_wait_asynccnt 0x8" ::: "memory");
#endif
#endif
}

// ---- fused cosine-similarity GEMM + streaming top-16 ----------------------
// One wave per 16-row stripe. B tiles double-buffered in LDS one iteration
// ahead (TDM/async), sorted top-16 (val,idx) kept in registers, fully
// unrolled; strict '>' compares reproduce lax.top_k tie order.

__global__ void __launch_bounds__(32) sim_topk_kernel(
    const __bf16* __restrict__ xn, int* __restrict__ knn, int n_total) {
  __shared__ __attribute__((aligned(32))) __bf16 btile[2][16 * DD];
  __shared__ float tile[16 * 17];
  int lane = threadIdx.x;
  int m = lane & 15, hi = lane >> 4;
  int n0 = blockIdx.x * 16;

  const __bf16* arow = xn + (size_t)(n0 + m) * DD;
  v16bf aF[4];
#pragma unroll
  for (int kc = 0; kc < 4; ++kc) aF[kc] = load_a_frag(arow, kc * 32, hi);

  float tv[16];
  int ti[16];
#pragma unroll
  for (int s = 0; s < 16; ++s) { tv[s] = -3.4e38f; ti[s] = 0; }

  int nct = n_total / 16;
  stage_tile(xn, 0, &btile[0][0], lane);            // prologue
  for (int ct = 0; ct < nct; ++ct) {
    int col0 = ct * 16;
    int nxt = (ct + 1 < nct) ? (ct + 1) * 16 : 0;   // wrap keeps wait invariant
    stage_tile(xn, nxt, &btile[(ct + 1) & 1][0], lane);
    __builtin_prefetch(xn + (size_t)(nxt + m) * DD, 0, 3);
    stage_wait();
    __syncthreads();

    const __bf16* bt = &btile[ct & 1][0];
    v16bf bF[4];
#pragma unroll
    for (int kc = 0; kc < 4; ++kc)   // column (lane&15)'s K-vector, contiguous
      bF[kc] = *(const v16bf*)(bt + m * DD + kc * 32 + hi * 16);
    v8f c = {};
#pragma unroll
    for (int kc = 0; kc < 4; ++kc)
      c = __builtin_amdgcn_wmma_f32_16x16x32_bf16(false, aF[kc], false, bF[kc],
                                                  (short)0, c, false, false);
#pragma unroll
    for (int v = 0; v < 8; ++v) tile[(hi * 8 + v) * 17 + m] = c[v];
    __syncthreads();
    if (lane < 16) {
#pragma unroll
      for (int cc = 0; cc < 16; ++cc) {
        float val = tile[lane * 17 + cc];
        int cidx = col0 + cc;
        if (val > tv[15]) {               // common case: one compare
          tv[15] = val; ti[15] = cidx;
#pragma unroll
          for (int p = 15; p > 0; --p) {  // single bubble pass
            if (tv[p] > tv[p - 1]) {
              float fv = tv[p]; tv[p] = tv[p - 1]; tv[p - 1] = fv;
              int iv = ti[p]; ti[p] = ti[p - 1]; ti[p - 1] = iv;
            }
          }
        }
      }
    }
    __syncthreads();
  }
  if (lane < 16) {
#pragma unroll
    for (int s = 0; s < 16; ++s) knn[(size_t)(n0 + lane) * KN_ + s] = ti[s];
  }
}

// ---- VertexConv (fused): WMMA mult -> softmax -> alpha -> alpha@x ---------
// out[n,d] = sum_j alpha[n,j]*x[n,j,d] + bK1,
// alpha[n,j] = sum_i wK1[i]*softmax_j(mult[n,i,:]). One wave per 16 vertices.

template <int K>
__global__ void __launch_bounds__(32) vconv_kernel(
    const float* __restrict__ feats, const __bf16* __restrict__ fbf,
    const int* __restrict__ idx, int idxStride,
    const __bf16* __restrict__ wkk_bf, const float* __restrict__ bkk,
    const float* __restrict__ wk1, const float* __restrict__ bk1,
    float* __restrict__ out, int outStride) {
  __shared__ float mult[16 * K];
  __shared__ float alpha_s[16 * K];
  int lane = threadIdx.x;
  int m = lane & 15, hi = lane >> 4;
  int n0 = blockIdx.x * 16;

  float alpha[K];
#pragma unroll
  for (int j = 0; j < K; ++j) alpha[j] = 0.f;

  for (int i = 0; i < K; ++i) {
    int r = idx[(size_t)(n0 + m) * idxStride + i];  // gathered A rows
    const __bf16* arow = fbf + (size_t)r * DD;
    v16bf aF[4];
#pragma unroll
    for (int kc = 0; kc < 4; ++kc) aF[kc] = load_a_frag(arow, kc * 32, hi);
#pragma unroll
    for (int jt = 0; jt < K / 16; ++jt) {
      int j0 = jt * 16;
      // B[k][j] = wKK[i][j0+j][k]: column j's K-vector is row (i*K+j0+j)
      const __bf16* brow = wkk_bf + (size_t)(i * K + j0 + m) * DD;
      v16bf bF[4];
#pragma unroll
      for (int kc = 0; kc < 4; ++kc) bF[kc] = load_b_frag(brow, kc * 32, hi);
      v8f c = {};
#pragma unroll
      for (int kc = 0; kc < 4; ++kc)
        c = __builtin_amdgcn_wmma_f32_16x16x32_bf16(false, aF[kc], false,
                                                    bF[kc], (short)0, c, false,
                                                    false);
#pragma unroll
      for (int v = 0; v < 8; ++v) mult[(hi * 8 + v) * K + j0 + m] = c[v];
    }
    __syncthreads();
    if (lane < 16) {  // softmax over j for this i, accumulate alpha
      float val[K];
      float mx = -3.4e38f;
#pragma unroll
      for (int j = 0; j < K; ++j) {
        val[j] = mult[lane * K + j] + bkk[i * K + j];
        mx = fmaxf(mx, val[j]);
      }
      float sum = 0.f;
#pragma unroll
      for (int j = 0; j < K; ++j) { val[j] = __expf(val[j] - mx); sum += val[j]; }
      float sc = wk1[i] / sum;
#pragma unroll
      for (int j = 0; j < K; ++j) alpha[j] += sc * val[j];
    }
    __syncthreads();
  }

  if (lane < 16) {
#pragma unroll
    for (int j = 0; j < K; ++j) alpha_s[lane * K + j] = alpha[j];
  }
  __syncthreads();

  float bias = bk1[0];
  for (int m2 = 0; m2 < 16; ++m2) {
    float a0 = bias, a1 = bias, a2 = bias, a3 = bias;
    for (int j = 0; j < K; ++j) {
      int rj = idx[(size_t)(n0 + m2) * idxStride + j];  // wave-uniform
      float a = alpha_s[m2 * K + j];
      const float* fr = feats + (size_t)rj * DD;        // coalesced fp32
      a0 += a * fr[lane];
      a1 += a * fr[lane + 32];
      a2 += a * fr[lane + 64];
      a3 += a * fr[lane + 96];
    }
    float* o = out + (size_t)(n0 + m2) * outStride;
    o[lane] = a0; o[lane + 32] = a1; o[lane + 64] = a2; o[lane + 96] = a3;
  }
}

// ---- EdgeConv attention: one wave per vertex, lanes = 32 hidden units -----

__global__ void __launch_bounds__(256) edge_attn_kernel(
    const float* __restrict__ xstack, const float* __restrict__ w1,
    const float* __restrict__ b1, const float* __restrict__ w2,
    const float* __restrict__ b2, __bf16* __restrict__ agg_bf, int n_total) {
  int lane = threadIdx.x & 31, wid = threadIdx.x >> 5;
  int n = blockIdx.x * 8 + wid;
  if (n >= n_total) return;
  const float* x = xstack + (size_t)n * 5 * DD;
  float sc[5];
#pragma unroll
  for (int t = 0; t < 5; ++t) {
    float h = b1[lane];
    for (int d = 0; d < DD; ++d) h += x[t * DD + d] * w1[d * 32 + lane];
    h = fmaxf(h, 0.f);
    float r = h * w2[lane];
#pragma unroll
    for (int off = 16; off; off >>= 1) r += __shfl_xor(r, off, 32);
    sc[t] = r + b2[0];
  }
  float mx = sc[0];
#pragma unroll
  for (int t = 1; t < 5; ++t) mx = fmaxf(mx, sc[t]);
  float sum = 0.f;
#pragma unroll
  for (int t = 0; t < 5; ++t) { sc[t] = __expf(sc[t] - mx); sum += sc[t]; }
  float inv = 1.f / sum;
#pragma unroll
  for (int t = 0; t < 5; ++t) sc[t] *= inv;
#pragma unroll
  for (int q = 0; q < 4; ++q) {
    int d = lane + 32 * q;
    float a = 0.f;
#pragma unroll
    for (int t = 0; t < 5; ++t) a += sc[t] * x[t * DD + d];
    agg_bf[(size_t)n * DD + d] = (__bf16)a;
  }
}

// ---- final FC: (N x 128) @ (128 x 128) via WMMA, + bias, ReLU -------------

__global__ void __launch_bounds__(32) final_fc_kernel(
    const __bf16* __restrict__ agg_bf, const __bf16* __restrict__ fcwT_bf,
    const float* __restrict__ fcb, float* __restrict__ out) {
  int lane = threadIdx.x;
  int m = lane & 15, hi = lane >> 4;
  int n0 = blockIdx.x * 16;
  int col0 = blockIdx.y * 16;
  const __bf16* arow = agg_bf + (size_t)(n0 + m) * DD;
  const __bf16* brow = fcwT_bf + (size_t)(col0 + m) * DD;  // pre-transposed
  v16bf aF[4], bF[4];
#pragma unroll
  for (int kc = 0; kc < 4; ++kc) {
    aF[kc] = load_a_frag(arow, kc * 32, hi);
    bF[kc] = load_b_frag(brow, kc * 32, hi);
  }
  v8f c = {};
#pragma unroll
  for (int kc = 0; kc < 4; ++kc)
    c = __builtin_amdgcn_wmma_f32_16x16x32_bf16(false, aF[kc], false, bF[kc],
                                                (short)0, c, false, false);
  float bias = fcb[col0 + m];
#pragma unroll
  for (int v = 0; v < 8; ++v) {
    int row = n0 + hi * 8 + v;
    out[(size_t)row * DD + col0 + m] = fmaxf(c[v] + bias, 0.f);
  }
}

// ---------------------------------------------------------------------------

extern "C" void kernel_launch(void* const* d_in, const int* in_sizes, int n_in,
                              void* d_out, int out_size, void* d_ws,
                              size_t ws_size, hipStream_t stream) {
  (void)in_sizes; (void)n_in; (void)out_size; (void)ws_size;
  const int N = NV;
  const float* feats       = (const float*)d_in[1];
  const int*   cluster_idx = (const int*)d_in[2];   // (N, 3, 16)
  const int*   struct_idx  = (const int*)d_in[3];   // (N, 32)
  const float* wKK_c = (const float*)d_in[5];
  const float* bKK_c = (const float*)d_in[6];
  const float* wK1_c = (const float*)d_in[7];
  const float* bK1_c = (const float*)d_in[8];
  const float* wKK_n = (const float*)d_in[9];
  const float* bKK_n = (const float*)d_in[10];
  const float* wK1_n = (const float*)d_in[11];
  const float* bK1_n = (const float*)d_in[12];
  const float* wKK_s = (const float*)d_in[13];
  const float* bKK_s = (const float*)d_in[14];
  const float* wK1_s = (const float*)d_in[15];
  const float* bK1_s = (const float*)d_in[16];
  const float* ec_w1 = (const float*)d_in[17];
  const float* ec_b1 = (const float*)d_in[18];
  const float* ec_w2 = (const float*)d_in[19];
  const float* ec_b2 = (const float*)d_in[20];
  const float* fc_w  = (const float*)d_in[21];
  const float* fc_b  = (const float*)d_in[22];

  char* w = (char*)d_ws;
  size_t off = 0;
  auto carve = [&](size_t bytes) {
    void* p = w + off;
    off = (off + bytes + 255) & ~(size_t)255;
    return p;
  };
  __bf16* xn_bf   = (__bf16*)carve((size_t)N * DD * 2);
  __bf16* f_bf    = (__bf16*)carve((size_t)N * DD * 2);
  int*    knn_idx = (int*)carve((size_t)N * KN_ * 4);
  __bf16* wc_bf   = (__bf16*)carve((size_t)KC_ * KC_ * DD * 2);
  __bf16* wn_bf   = (__bf16*)carve((size_t)KN_ * KN_ * DD * 2);
  __bf16* wsb_bf  = (__bf16*)carve((size_t)KS_ * KS_ * DD * 2);
  __bf16* fcwT_bf = (__bf16*)carve((size_t)DD * DD * 2);
  float*  xstack  = (float*)carve((size_t)N * 5 * DD * 4);
  __bf16* agg_bf  = (__bf16*)carve((size_t)N * DD * 2);

  prep_norm_kernel<<<(N + 7) / 8, 256, 0, stream>>>(feats, xn_bf, f_bf, N);
  prep_weights_kernel<<<64, 256, 0, stream>>>(wKK_c, wKK_n, wKK_s, fc_w, wc_bf,
                                              wn_bf, wsb_bf, fcwT_bf);

  sim_topk_kernel<<<N / 16, 32, 0, stream>>>(xn_bf, knn_idx, N);

  for (int c = 0; c < 3; ++c) {   // cluster hyperedges t=0..2
    vconv_kernel<KC_><<<N / 16, 32, 0, stream>>>(
        feats, f_bf, cluster_idx + c * KC_, 3 * KC_, wc_bf, bKK_c, wK1_c, bK1_c,
        xstack + (size_t)c * DD, 5 * DD);
  }
  vconv_kernel<KN_><<<N / 16, 32, 0, stream>>>(   // knn hyperedge t=3
      feats, f_bf, knn_idx, KN_, wn_bf, bKK_n, wK1_n, bK1_n,
      xstack + (size_t)3 * DD, 5 * DD);
  vconv_kernel<KS_><<<N / 16, 32, 0, stream>>>(   // structure hyperedge t=4
      feats, f_bf, struct_idx, KS_, wsb_bf, bKK_s, wK1_s, bK1_s,
      xstack + (size_t)4 * DD, 5 * DD);

  edge_attn_kernel<<<(N + 7) / 8, 256, 0, stream>>>(xstack, ec_w1, ec_b1, ec_w2,
                                                    ec_b2, agg_bf, N);

  final_fc_kernel<<<dim3(N / 16, DD / 16), 32, 0, stream>>>(agg_bf, fcwT_bf,
                                                            fc_b, (float*)d_out);
}
